// LinearClassifier2_35296041238606
// MI455X (gfx1250) — compile-verified, matrix operand order checked by reading, not run on hardware
//
#include <hip/hip_runtime.h>

#define NODES 3249      // 57*57
#define MPAD  3264      // 204*16
#define CIN   768
#define C8    96
#define LOUT  21
#define OUTHW 800
#define PIX   640000
#define NMASK 64
#define ZSTRIDE 32      // padded channel stride for z

typedef __attribute__((ext_vector_type(16))) _Float16 v16h;
typedef __attribute__((ext_vector_type(8)))  _Float16 v8h;
typedef __attribute__((ext_vector_type(8)))  float    v8f;

__device__ __forceinline__ v8h cvt_pack8(float4 u, float4 v) {
  v8h r;
  r[0] = (_Float16)u.x; r[1] = (_Float16)u.y; r[2] = (_Float16)u.z; r[3] = (_Float16)u.w;
  r[4] = (_Float16)v.x; r[5] = (_Float16)v.y; r[6] = (_Float16)v.z; r[7] = (_Float16)v.w;
  return r;
}

#define SHUF16(a, b) __builtin_shufflevector(a, b, 0,1,2,3,4,5,6,7,8,9,10,11,12,13,14,15)

// ---------------------------------------------------------------------------
// Kernel 1: y = relu(emb @ W1^T + b1)   [B, NODES, 96]  f32 -> f16 WMMA
// grid(204, B), block(192) = 6 waves; wave w owns n-tile w (16 cols), K=768.
// Out-of-range M rows are CLAMPED (not zeroed): row m of A only affects row m
// of D, and the epilogue guards stores, so no EXEC juggling in the K loop.
// ---------------------------------------------------------------------------
__global__ void __launch_bounds__(192)
conv1_wmma(const float* __restrict__ emb, const float* __restrict__ w1,
           const float* __restrict__ b1, _Float16* __restrict__ y) {
  const int lane = threadIdx.x & 31;
  const int wave = threadIdx.x >> 5;     // 0..5
  const int ll   = lane & 15;
  const int hi   = lane >> 4;            // 0 or 1
  const int m0   = blockIdx.x * 16;
  const int n0   = wave * 16;
  const int b    = blockIdx.y;

  int arow = m0 + ll;
  if (arow > NODES - 1) arow = NODES - 1;               // clamp, not zero
  const float* ap = emb + ((size_t)b * NODES + arow) * CIN;
  const float* bp = w1 + (size_t)(n0 + ll) * CIN;       // col < 96 always

  v8f acc = {};
  #pragma unroll 4
  for (int ks = 0; ks < CIN / 32; ++ks) {
    const int kb = ks * 32 + hi * 8;
    const float4* pa0 = (const float4*)(ap + kb);
    const float4* pa1 = (const float4*)(ap + kb + 16);
    v16h A = SHUF16(cvt_pack8(pa0[0], pa0[1]), cvt_pack8(pa1[0], pa1[1]));
    const float4* pb0 = (const float4*)(bp + kb);
    const float4* pb1 = (const float4*)(bp + kb + 16);
    v16h Bf = SHUF16(cvt_pack8(pb0[0], pb0[1]), cvt_pack8(pb1[0], pb1[1]));
    acc = __builtin_amdgcn_wmma_f32_16x16x32_f16(false, A, false, Bf,
                                                 (short)0, acc, false, false);
  }
  const float bias = b1[n0 + ll];
  #pragma unroll
  for (int g = 0; g < 8; ++g) {
    const int row = m0 + g + hi * 8;           // D: vgpr g -> M=g (lanes<16), g+8
    if (row < NODES) {
      float v = acc[g] + bias;
      v = v > 0.0f ? v : 0.0f;
      y[((size_t)b * MPAD + row) * C8 + (n0 + ll)] = (_Float16)v;
    }
  }
}

// ---------------------------------------------------------------------------
// Kernel 2: z = y @ W2^T  (no bias; bias folded into means at finalize)
// grid(204, B), block(64) = 2 waves, each wave one n-tile (N padded 21->32).
// A rows clamped to NODES-1 (valid y data); B cols clamped to 0: pad columns
// of z get deterministic garbage that is never read (only o<21 consumed).
// z stored node-interleaved [B][MPAD][32] f32 so D-frag stores coalesce.
// ---------------------------------------------------------------------------
__global__ void __launch_bounds__(64)
conv2_wmma(const _Float16* __restrict__ y, const float* __restrict__ w2,
           float* __restrict__ z) {
  const int lane = threadIdx.x & 31;
  const int wave = threadIdx.x >> 5;     // 0..1
  const int ll   = lane & 15;
  const int hi   = lane >> 4;
  const int m0   = blockIdx.x * 16;
  const int n0   = wave * 16;
  const int b    = blockIdx.y;

  int arow = m0 + ll;
  if (arow > NODES - 1) arow = NODES - 1;               // clamp
  const _Float16* ap = y + ((size_t)b * MPAD + arow) * C8;
  int bcol = n0 + ll;
  if (bcol > LOUT - 1) bcol = 0;                        // clamp pad cols
  const float* bp = w2 + (size_t)bcol * C8;

  v8f acc = {};
  #pragma unroll
  for (int ks = 0; ks < C8 / 32; ++ks) {
    const int kb = ks * 32 + hi * 8;
    v8h lo  = *(const v8h*)(ap + kb);        // 8 f16 = 16B, aligned
    v8h hi8 = *(const v8h*)(ap + kb + 16);
    v16h A = SHUF16(lo, hi8);
    const float4* pb0 = (const float4*)(bp + kb);
    const float4* pb1 = (const float4*)(bp + kb + 16);
    v16h Bf = SHUF16(cvt_pack8(pb0[0], pb0[1]), cvt_pack8(pb1[0], pb1[1]));
    acc = __builtin_amdgcn_wmma_f32_16x16x32_f16(false, A, false, Bf,
                                                 (short)0, acc, false, false);
  }
  #pragma unroll
  for (int g = 0; g < 8; ++g) {
    const int row = m0 + g + hi * 8;           // < MPAD always
    z[((size_t)b * MPAD + row) * ZSTRIDE + (n0 + ll)] = acc[g];
  }
}

// ---------------------------------------------------------------------------
// Bilinear (half-pixel, edge clamp) of 21 channels from z at (h,w).
// ---------------------------------------------------------------------------
__device__ __forceinline__ void bilinear21(const float* __restrict__ zb,
                                           int h, int w, float* __restrict__ r) {
  const float SC = 57.0f / 800.0f;
  const float sy = ((float)h + 0.5f) * SC - 0.5f;
  const float sx = ((float)w + 0.5f) * SC - 0.5f;
  const float fly = floorf(sy), flx = floorf(sx);
  const float fy = sy - fly,    fx = sx - flx;
  int iy = (int)fly, ix = (int)flx;
  const int y0 = iy     < 0 ? 0 : (iy     > 56 ? 56 : iy);
  const int y1 = iy + 1 < 0 ? 0 : (iy + 1 > 56 ? 56 : iy + 1);
  const int x0 = ix     < 0 ? 0 : (ix     > 56 ? 56 : ix);
  const int x1 = ix + 1 < 0 ? 0 : (ix + 1 > 56 ? 56 : ix + 1);
  const float* p00 = zb + (size_t)(y0 * 57 + x0) * ZSTRIDE;
  const float* p01 = zb + (size_t)(y0 * 57 + x1) * ZSTRIDE;
  const float* p10 = zb + (size_t)(y1 * 57 + x0) * ZSTRIDE;
  const float* p11 = zb + (size_t)(y1 * 57 + x1) * ZSTRIDE;
  const float w00 = (1.0f - fy) * (1.0f - fx), w01 = (1.0f - fy) * fx;
  const float w10 = fy * (1.0f - fx),          w11 = fy * fx;
  #pragma unroll
  for (int o = 0; o < LOUT; ++o)
    r[o] = w00 * p00[o] + w01 * p01[o] + w10 * p10[o] + w11 * p11[o];
}

// ---------------------------------------------------------------------------
// Kernel 0: zero the segment accumulators (every call -> deterministic).
// ---------------------------------------------------------------------------
__global__ void __launch_bounds__(256)
zero_acc(float* __restrict__ S, unsigned* __restrict__ CNT, int B) {
  const int i = blockIdx.x * blockDim.x + threadIdx.x;
  if (i < B * NMASK * LOUT) S[i] = 0.0f;
  if (i < B * NMASK)        CNT[i] = 0u;
}

// ---------------------------------------------------------------------------
// Kernel 3: per-segment sums of r (recomputed bilinear) via LDS atomics.
// stride 21 vs 64 banks: gcd(21,64)=1 -> distinct segs hit distinct banks.
// ---------------------------------------------------------------------------
__global__ void __launch_bounds__(256)
segsum_kernel(const float* __restrict__ z, const int* __restrict__ labels,
              float* __restrict__ S, unsigned* __restrict__ CNT) {
  __shared__ float    ssum[NMASK * LOUT];
  __shared__ unsigned scnt[NMASK];
  const int b = blockIdx.y;
  for (int i = threadIdx.x; i < NMASK * LOUT; i += blockDim.x) ssum[i] = 0.0f;
  for (int i = threadIdx.x; i < NMASK;        i += blockDim.x) scnt[i] = 0u;
  __syncthreads();

  const float* zb = z + (size_t)b * MPAD * ZSTRIDE;
  const int*   lb = labels + (size_t)b * PIX;
  for (int p = blockIdx.x * blockDim.x + threadIdx.x; p < PIX;
       p += gridDim.x * blockDim.x) {
    const int h = p / OUTHW, w = p - h * OUTHW;
    float r[LOUT];
    bilinear21(zb, h, w, r);
    const int m = __builtin_nontemporal_load(lb + p) & (NMASK - 1);
    atomicAdd(&scnt[m], 1u);
    #pragma unroll
    for (int o = 0; o < LOUT; ++o) atomicAdd(&ssum[m * LOUT + o], r[o]);
  }
  __syncthreads();
  for (int i = threadIdx.x; i < NMASK * LOUT; i += blockDim.x)
    atomicAdd(&S[(size_t)b * NMASK * LOUT + i], ssum[i]);
  for (int i = threadIdx.x; i < NMASK; i += blockDim.x)
    if (scnt[i]) atomicAdd(&CNT[(size_t)b * NMASK + i], scnt[i]);
}

// ---------------------------------------------------------------------------
// Kernel 4: out[b,o,h,w] = r + S[seg]/max(cnt,1) + b2[o]  (means+bias in LDS)
// ---------------------------------------------------------------------------
__global__ void __launch_bounds__(256)
final_kernel(const float* __restrict__ z, const int* __restrict__ labels,
             const float* __restrict__ S, const unsigned* __restrict__ CNT,
             const float* __restrict__ b2, float* __restrict__ out) {
  __shared__ float mean[NMASK * LOUT];
  const int b = blockIdx.y;
  for (int i = threadIdx.x; i < NMASK * LOUT; i += blockDim.x) {
    const int m = i / LOUT, o = i - m * LOUT;
    const unsigned c = CNT[(size_t)b * NMASK + m];
    const float inv = 1.0f / (float)(c > 0u ? c : 1u);
    mean[i] = S[(size_t)b * NMASK * LOUT + i] * inv + b2[o];
  }
  __syncthreads();

  const float* zb = z + (size_t)b * MPAD * ZSTRIDE;
  const int*   lb = labels + (size_t)b * PIX;
  float*       ob = out + (size_t)b * LOUT * PIX;
  for (int p = blockIdx.x * blockDim.x + threadIdx.x; p < PIX;
       p += gridDim.x * blockDim.x) {
    const int h = p / OUTHW, w = p - h * OUTHW;
    float r[LOUT];
    bilinear21(zb, h, w, r);
    const int m = __builtin_nontemporal_load(lb + p) & (NMASK - 1);
    const float* mm = &mean[m * LOUT];
    #pragma unroll
    for (int o = 0; o < LOUT; ++o)
      __builtin_nontemporal_store(r[o] + mm[o], ob + (size_t)o * PIX + p);
  }
}

// ---------------------------------------------------------------------------
extern "C" void kernel_launch(void* const* d_in, const int* in_sizes, int n_in,
                              void* d_out, int out_size, void* d_ws, size_t ws_size,
                              hipStream_t stream) {
  const float* emb    = (const float*)d_in[0];
  const int*   labels = (const int*)  d_in[1];
  const float* w1     = (const float*)d_in[2];
  const float* b1     = (const float*)d_in[3];
  const float* w2     = (const float*)d_in[4];
  const float* b2     = (const float*)d_in[5];
  float*       out    = (float*)d_out;

  const int B = in_sizes[0] / (NODES * CIN);

  // workspace layout (all offsets 256B-aligned for B=2)
  char* ws = (char*)d_ws;
  _Float16* y_ws = (_Float16*)ws;
  size_t off = (size_t)B * MPAD * C8 * sizeof(_Float16);
  float* z_ws = (float*)(ws + off);
  off += (size_t)B * MPAD * ZSTRIDE * sizeof(float);
  float* S = (float*)(ws + off);
  off += (size_t)B * NMASK * LOUT * sizeof(float);
  unsigned* CNT = (unsigned*)(ws + off);

  const int mtiles = (NODES + 15) / 16;   // 204

  const int nzero = B * NMASK * LOUT;
  zero_acc<<<(nzero + 255) / 256, 256, 0, stream>>>(S, CNT, B);
  conv1_wmma<<<dim3(mtiles, B), 192, 0, stream>>>(emb, w1, b1, y_ws);
  conv2_wmma<<<dim3(mtiles, B), 64, 0, stream>>>(y_ws, w2, z_ws);
  segsum_kernel<<<dim3(256, B), 256, 0, stream>>>(z_ws, labels, S, CNT);
  final_kernel<<<dim3(1000, B), 256, 0, stream>>>(z_ws, labels, S, CNT, b2, out);
}
